// MultiHeadAttention_30485677867992
// MI455X (gfx1250) — compile-verified, matrix operand order checked by reading, not run on hardware
//
#include <hip/hip_runtime.h>
#include <hip/hip_bf16.h>

// ---------------------------------------------------------------------------
// MultiHeadAttention forward for MI455X (gfx1250).
// f16 WMMA w/ f32 accumulation; TDM (tensor_load_to_lds) staged, double-
// buffered GEMMs; flash-style attention.
// BS=2, QLEN=2048, DIM=2048, H=16, DH=128
// ---------------------------------------------------------------------------

typedef __attribute__((ext_vector_type(16))) _Float16 v16h;
typedef __attribute__((ext_vector_type(8)))  _Float16 v8h;
typedef __attribute__((ext_vector_type(8)))  float    v8f;
typedef __attribute__((ext_vector_type(4)))  unsigned int u32x4;
typedef __attribute__((ext_vector_type(8)))  int      i32x8;
typedef __attribute__((ext_vector_type(4)))  int      i32x4;

#define BSZ   2
#define QL    2048
#define DIMC  2048
#define NH    16
#define DH    128
#define ROWS  (BSZ * QL)      // 4096 tokens
#define GK    DIMC            // GEMM reduction dim
#define GN    DIMC            // GEMM output dim
#define QSCALE 0.08838834764831845f   // 1/sqrt(128)

enum { MODE_Q = 0, MODE_K = 1, MODE_V = 2, MODE_OUT = 3 };

// ---------------------------------------------------------------------------
// WMMA helper:  D(16x16,f32) = A(16x32,f16) * B(32x16,f16) + C
// ---------------------------------------------------------------------------
__device__ __forceinline__ v8f wmma_f16(v16h a, v16h b, v8f c) {
  return __builtin_amdgcn_wmma_f32_16x16x32_f16(
      /*neg_a=*/false, a, /*neg_b=*/false, b,
      /*c_mod=*/(short)0, c, /*reuse_a=*/false, /*reuse_b=*/false);
}

// Load a 16x32 f16 fragment (A layout; also used for B where column n of B is
// contiguous in memory). p points to element (0,0); ld = row stride in halves.
__device__ __forceinline__ v16h frag_ld(const _Float16* p, int ld) {
  const int lane = threadIdx.x & 31;
  const int r  = lane & 15;
  const int kb = (lane >> 4) << 3;            // 0 or 8
  const _Float16* q = p + (size_t)r * ld + kb;
  v8h lo = *(const v8h*)(q);
  v8h hi = *(const v8h*)(q + 16);
  v16h out;
#pragma unroll
  for (int i = 0; i < 8; ++i) { out[i] = lo[i]; out[i + 8] = hi[i]; }
  return out;
}

__device__ __forceinline__ v8h cvt8(float4 f0, float4 f1) {
  v8h h;
  h[0] = (_Float16)f0.x; h[1] = (_Float16)f0.y;
  h[2] = (_Float16)f0.z; h[3] = (_Float16)f0.w;
  h[4] = (_Float16)f1.x; h[5] = (_Float16)f1.y;
  h[6] = (_Float16)f1.z; h[7] = (_Float16)f1.w;
  return h;
}

// ---------------------------------------------------------------------------
// Tensor Data Mover: DMA a [rows x 32-half] f16 tile (row stride `ld` halves)
// from global memory into LDS, writing padded 40-half rows (pad 4 DWORDs after
// every 16 DWORDs) so fragment reads are bank-conflict free.
// Descriptor layout per CDNA5 ISA.
// ---------------------------------------------------------------------------
__device__ __forceinline__ void tdm_tile(const _Float16* gptr, int ld, int rows,
                                         unsigned lds_byte_addr) {
  const unsigned long long ga = (unsigned long long)(size_t)gptr;

  u32x4 g0;
  g0[0] = 1u;                                   // count=1 (valid), load, no gather
  g0[1] = lds_byte_addr;                        // LDS dest (bytes)
  g0[2] = (unsigned)ga;                         // global_addr[31:0]
  g0[3] = ((unsigned)(ga >> 32) & 0x1FFFFFFu)   // global_addr[56:32]
        | (2u << 30);                           // type = 2 ("image")

  i32x8 g1;
  g1[0] = (1 << 16)      // data_size = 2 bytes
        | (1 << 20)      // pad_enable
        | (3 << 22)      // pad_interval: 16 DWORDs (= 32 halves = tile row)
        | (3 << 25);     // pad_amount:   4 DWORDs (= 8 halves)
  g1[1] = (ld & 0xFFFF) << 16;                          // tensor_dim0[15:0]
  g1[2] = ((ld >> 16) & 0xFFFF) | ((rows & 0xFFFF) << 16); // dim0 hi | dim1 lo
  g1[3] = ((rows >> 16) & 0xFFFF) | (32 << 16);         // dim1 hi | tile_dim0=32
  g1[4] = (rows & 0xFFFF);                              // tile_dim1=rows, tile_dim2=0
  g1[5] = ld;                                           // tensor_dim0_stride[31:0]
  g1[6] = 0;                                            // stride hi, dim1_stride lo
  g1[7] = 0;

  i32x4 z4 = {0, 0, 0, 0};
#if defined(__clang_major__) && (__clang_major__ >= 23)
  i32x8 z8 = {0, 0, 0, 0, 0, 0, 0, 0};
  __builtin_amdgcn_tensor_load_to_lds(g0, g1, z4, z4, z8, 0);
#else
  __builtin_amdgcn_tensor_load_to_lds(g0, g1, z4, z4, 0);
#endif
}

// Low 32 bits of a generic pointer into the LDS aperture = wave-relative
// LDS byte offset (ISA 07_vmem §10.2).
__device__ __forceinline__ unsigned lds_of(const void* p) {
  return (unsigned)(size_t)p;
}

// ---------------------------------------------------------------------------
// f32 -> f16 conversion (one-time; 8 elements/thread)
// ---------------------------------------------------------------------------
__global__ __launch_bounds__(256)
void cvt_f16_kernel(const float* __restrict__ src, _Float16* __restrict__ dst,
                    int n8) {
  const int i = blockIdx.x * 256 + threadIdx.x;
  if (i < n8) {
    const float* g = src + (size_t)i * 8;
    float4 f0 = *(const float4*)(g);
    float4 f1 = *(const float4*)(g + 4);
    *(v8h*)(dst + (size_t)i * 8) = cvt8(f0, f1);
  }
}

// ---------------------------------------------------------------------------
// TDM-staged, double-buffered GEMM:
//   out = A[ROWS x GK](f16) @ W[GN x GK](f16)^T + bias   (per-mode epilogue)
// Block: 256 threads = 8 waves. Block tile 64(M) x 256(N), K-step 32.
// Wave 0 drives the TDM pipeline; every wave does 8 WMMA per K-step.
// ---------------------------------------------------------------------------
template <int MODE>
__global__ __launch_bounds__(256)
void gemm_wmma_kernel(const _Float16* __restrict__ A,
                      const _Float16* __restrict__ W,
                      const float* __restrict__ bias, void* __restrict__ outp) {
  __shared__ _Float16 lA[2][64][40];     // ping-pong, padded 80B rows
  __shared__ _Float16 lB[2][256][40];

  const int t    = threadIdx.x;
  const int lane = t & 31;
  const int w    = t >> 5;       // wave 0..7
  const int wm   = w >> 2;       // 0..1
  const int wn   = w & 3;        // 0..3
  const int row0 = blockIdx.y * 64;
  const int n0   = blockIdx.x * 256;

  v8f c[2][4] = {};

  // Prime the pipeline: DMA tile 0 into buffer 0.
  if (w == 0) {
    tdm_tile(A + (size_t)row0 * GK, GK, 64,  lds_of(&lA[0][0][0]));
    tdm_tile(W + (size_t)n0  * GK, GK, 256, lds_of(&lB[0][0][0]));
  }

  for (int k0 = 0; k0 < GK; k0 += 32) {
    const int cur = (k0 >> 5) & 1;
    if (w == 0) {
      if (k0 + 32 < GK) {
        // Issue next tile's DMA into the other buffer, then wait until only
        // those 2 ops remain outstanding (=> current buffer complete).
        tdm_tile(A + (size_t)row0 * GK + k0 + 32, GK, 64,
                 lds_of(&lA[cur ^ 1][0][0]));
        tdm_tile(W + (size_t)n0 * GK + k0 + 32, GK, 256,
                 lds_of(&lB[cur ^ 1][0][0]));
        __builtin_amdgcn_s_wait_tensorcnt(2);
      } else {
        __builtin_amdgcn_s_wait_tensorcnt(0);
      }
    }
    __syncthreads();

    // --- compute on current buffer ---
    v16h a0 = frag_ld(&lA[cur][wm * 32 + 0][0], 40);
    v16h a1 = frag_ld(&lA[cur][wm * 32 + 16][0], 40);
#pragma unroll
    for (int j = 0; j < 4; ++j) {
      v16h b = frag_ld(&lB[cur][wn * 64 + j * 16][0], 40);
      c[0][j] = wmma_f16(a0, b, c[0][j]);
      c[1][j] = wmma_f16(a1, b, c[1][j]);
    }
    __syncthreads();   // all reads of `cur` done before it is re-DMA'd
  }

  // --- epilogue ---
  const int hl = lane >> 4;
  const int nl = lane & 15;
#pragma unroll
  for (int i = 0; i < 2; ++i) {
#pragma unroll
    for (int j = 0; j < 4; ++j) {
#pragma unroll
      for (int r = 0; r < 8; ++r) {
        const int m = row0 + wm * 32 + i * 16 + r + 8 * hl;
        const int n = n0 + wn * 64 + j * 16 + nl;
        const float val = c[i][j][r] + bias[n];
        if (MODE == MODE_Q) {
          ((_Float16*)outp)[(size_t)m * GN + n] = (_Float16)(val * QSCALE);
        } else if (MODE == MODE_K) {
          ((_Float16*)outp)[(size_t)m * GN + n] = (_Float16)val;
        } else if (MODE == MODE_V) {
          // transpose-store V per head: Vt[b, h, d, q]
          const int b  = m >> 11;       // / QL
          const int qq = m & (QL - 1);
          const int h  = n >> 7;        // / DH
          const int d  = n & (DH - 1);
          ((_Float16*)outp)[((size_t)((b * NH + h) * DH + d) << 11) + qq] =
              (_Float16)val;
        } else {  // MODE_OUT: f32
          ((float*)outp)[(size_t)m * GN + n] = val;
        }
      }
    }
  }
}

// ---------------------------------------------------------------------------
// Flash attention: one wave owns 16 queries x DH=128 of one (b,h).
// Streams keys in blocks of 32: 8 WMMA (scores) + online softmax + 8 WMMA (PV).
// ---------------------------------------------------------------------------
__global__ __launch_bounds__(256)
void attn_kernel(const _Float16* __restrict__ Q, const _Float16* __restrict__ Km,
                 const _Float16* __restrict__ Vt, const int* __restrict__ mask,
                 _Float16* __restrict__ Ctx) {
  __shared__ _Float16 lp[8][16][40];   // P tile staging per wave

  const int t    = threadIdx.x;
  const int lane = t & 31;
  const int w    = t >> 5;
  const int bh   = blockIdx.x;         // 0..BSZ*NH-1
  const int b    = bh >> 4;
  const int h    = bh & 15;
  const int q0   = (blockIdx.y * 8 + w) * 16;

  const size_t tok0 = (size_t)b * QL;
  const _Float16* Qp = Q + (tok0 + q0) * DIMC + h * DH;
  const _Float16* Kp = Km + tok0 * DIMC + h * DH;
  const _Float16* Vp = Vt + (size_t)(b * NH + h) * DH * QL;
  const int* mp = mask + b * QL;

  v16h qa[4];
#pragma unroll
  for (int t4 = 0; t4 < 4; ++t4) qa[t4] = frag_ld(Qp + t4 * 32, DIMC);

  v8f acc[8] = {};
  float mrun[8], lrun[8];
#pragma unroll
  for (int r = 0; r < 8; ++r) { mrun[r] = -3.0e38f; lrun[r] = 0.0f; }

  const int nl = lane & 15;
  const int hl = lane >> 4;

  for (int kc = 0; kc < QL; kc += 32) {
    // ---- scores S[16x32] = Q * K^T  (Q pre-scaled by 1/sqrt(DH)) ----
    v8f s0 = {}, s1 = {};
#pragma unroll
    for (int t4 = 0; t4 < 4; ++t4) {
      v16h kb0 = frag_ld(Kp + (size_t)kc * DIMC + t4 * 32, DIMC);
      v16h kb1 = frag_ld(Kp + (size_t)(kc + 16) * DIMC + t4 * 32, DIMC);
      s0 = wmma_f16(qa[t4], kb0, s0);
      s1 = wmma_f16(qa[t4], kb1, s1);
    }
    // ---- mask ----
    if (mp[kc + nl] == 0) {
#pragma unroll
      for (int r = 0; r < 8; ++r) s0[r] = -3.0e38f;
    }
    if (mp[kc + 16 + nl] == 0) {
#pragma unroll
      for (int r = 0; r < 8; ++r) s1[r] = -3.0e38f;
    }
    // ---- online softmax (row reductions within 16-lane halves) ----
#pragma unroll
    for (int r = 0; r < 8; ++r) {
      float v = fmaxf(s0[r], s1[r]);
      v = fmaxf(v, __shfl_xor(v, 1));
      v = fmaxf(v, __shfl_xor(v, 2));
      v = fmaxf(v, __shfl_xor(v, 4));
      v = fmaxf(v, __shfl_xor(v, 8));
      const float nm = fmaxf(mrun[r], v);
      const float alpha = __expf(mrun[r] - nm);
      mrun[r] = nm;
      const float p0 = __expf(s0[r] - nm);
      const float p1 = __expf(s1[r] - nm);
      float rs = p0 + p1;
      rs += __shfl_xor(rs, 1);
      rs += __shfl_xor(rs, 2);
      rs += __shfl_xor(rs, 4);
      rs += __shfl_xor(rs, 8);
      lrun[r] = lrun[r] * alpha + rs;
#pragma unroll
      for (int dd = 0; dd < 8; ++dd) acc[dd][r] *= alpha;
      const int mrow = r + 8 * hl;
      lp[w][mrow][nl]      = (_Float16)p0;
      lp[w][mrow][16 + nl] = (_Float16)p1;
    }
    // wave-private LDS region: only intra-wave store->load ordering needed
    asm volatile("s_wait_dscnt 0" ::: "memory");

    // ---- context += P(16x32) @ V(32x128) ----
    v16h pa = frag_ld(&lp[w][0][0], 40);
#pragma unroll
    for (int dd = 0; dd < 8; ++dd) {
      v16h vb = frag_ld(Vp + (size_t)(dd * 16) * QL + kc, QL);
      acc[dd] = wmma_f16(pa, vb, acc[dd]);
    }
  }

  // ---- normalize and store context (f16, [token, dim] layout) ----
  float inv[8];
#pragma unroll
  for (int r = 0; r < 8; ++r) inv[r] = lrun[r] > 0.0f ? 1.0f / lrun[r] : 0.0f;

  _Float16* cp = Ctx + (tok0 + q0) * DIMC + h * DH;
#pragma unroll
  for (int dd = 0; dd < 8; ++dd) {
#pragma unroll
    for (int r = 0; r < 8; ++r) {
      const int m = r + 8 * hl;
      const int d = dd * 16 + nl;
      cp[(size_t)m * DIMC + d] = (_Float16)(acc[dd][r] * inv[r]);
    }
  }
}

// ---------------------------------------------------------------------------
extern "C" void kernel_launch(void* const* d_in, const int* in_sizes, int n_in,
                              void* d_out, int out_size, void* d_ws, size_t ws_size,
                              hipStream_t stream) {
  const float* input = (const float*)d_in[0];
  const int*   mask  = (const int*)d_in[1];
  const float* wq    = (const float*)d_in[2];
  const float* bq    = (const float*)d_in[3];
  const float* wk    = (const float*)d_in[4];
  const float* bk    = (const float*)d_in[5];
  const float* wv    = (const float*)d_in[6];
  const float* bv    = (const float*)d_in[7];
  const float* wo    = (const float*)d_in[8];
  const float* bo    = (const float*)d_in[9];
  float* out = (float*)d_out;

  const size_t act = (size_t)ROWS * GN;   // 8M halves
  const size_t wm_ = (size_t)GN * GK;     // 4M halves

  _Float16* X16  = (_Float16*)d_ws;       // f16 copies
  _Float16* Wq16 = X16 + act;
  _Float16* Wk16 = Wq16 + wm_;
  _Float16* Wv16 = Wk16 + wm_;
  _Float16* Wo16 = Wv16 + wm_;
  _Float16* Q16  = Wo16 + wm_;            // intermediates
  _Float16* K16  = Q16 + act;
  _Float16* Vt   = K16 + act;
  _Float16* Ctx  = Vt + act;

  // One-time f32 -> f16 conversions (bandwidth trivial vs GEMM compute).
  cvt_f16_kernel<<<(int)(act / (256 * 8)), 256, 0, stream>>>(input, X16, (int)(act / 8));
  cvt_f16_kernel<<<(int)(wm_ / (256 * 8)), 256, 0, stream>>>(wq, Wq16, (int)(wm_ / 8));
  cvt_f16_kernel<<<(int)(wm_ / (256 * 8)), 256, 0, stream>>>(wk, Wk16, (int)(wm_ / 8));
  cvt_f16_kernel<<<(int)(wm_ / (256 * 8)), 256, 0, stream>>>(wv, Wv16, (int)(wm_ / 8));
  cvt_f16_kernel<<<(int)(wm_ / (256 * 8)), 256, 0, stream>>>(wo, Wo16, (int)(wm_ / 8));

  dim3 gg(GN / 256, ROWS / 64);
  gemm_wmma_kernel<MODE_Q><<<gg, 256, 0, stream>>>(X16, Wq16, bq, Q16);
  gemm_wmma_kernel<MODE_K><<<gg, 256, 0, stream>>>(X16, Wk16, bk, K16);
  gemm_wmma_kernel<MODE_V><<<gg, 256, 0, stream>>>(X16, Wv16, bv, Vt);

  attn_kernel<<<dim3(BSZ * NH, QL / 128), 256, 0, stream>>>(Q16, K16, Vt, mask, Ctx);

  gemm_wmma_kernel<MODE_OUT><<<gg, 256, 0, stream>>>(Ctx, Wo16, bo, out);
}
